// Head_70866960384064
// MI455X (gfx1250) — matrix operand order repaired and checked by reading, not compiled, not run on hardware
//
#include <hip/hip_runtime.h>
#include <hip/hip_bf16.h>

// ---------------------------------------------------------------------------
// Single-head attention (projection + causal flash attention), CDNA5 gfx1250.
// B=8, T=2048, C=1024, HS=HO=128. f16 WMMA (v_wmma_f32_16x16x32_f16), f32 acc.
// K/V tiles staged with GLOBAL_LOAD_ASYNC_TO_LDS_B128, double-buffered on
// ASYNCcnt so the DMA of block kb+1 overlaps the WMMAs of block kb.
// ---------------------------------------------------------------------------

typedef __attribute__((ext_vector_type(16))) _Float16 v16h;
typedef __attribute__((ext_vector_type(8)))  _Float16 v8h;
typedef __attribute__((ext_vector_type(2)))  __fp16   v2hp;  // cvt_pkrtz return type
typedef __attribute__((ext_vector_type(8)))  float    v8f;
typedef int v4i32 __attribute__((vector_size(16)));   // matches builtin param type

#define BATCH 8
#define TLEN  2048
#define CDIM  1024
#define HSD   128

#define QBLK  128
#define KBLK  64

#if defined(__has_builtin)
#if __has_builtin(__builtin_amdgcn_global_load_async_to_lds_b128)
#define HAVE_ASYNC_LDS 1
#endif
#endif
#ifndef HAVE_ASYNC_LDS
#define HAVE_ASYNC_LDS 0
#endif

#if HAVE_ASYNC_LDS
// Address-space casts via integer round-trip: flat->global is identity; flat
// LDS addr low 32 bits are the LDS offset.
static __device__ inline void async_copy16(const _Float16* g, _Float16* l) {
    __builtin_amdgcn_global_load_async_to_lds_b128(
        (__attribute__((address_space(1))) v4i32*)(unsigned long long)g,
        (__attribute__((address_space(3))) v4i32*)(unsigned int)(unsigned long long)l,
        /*imm offset*/0, /*cpol*/0);
}
#endif

static __device__ inline void wait_async_le0() {
#if HAVE_ASYNC_LDS
#if __has_builtin(__builtin_amdgcn_s_wait_asynccnt)
    __builtin_amdgcn_s_wait_asynccnt(0);
#else
    asm volatile("s_wait_asynccnt 0" ::: "memory");
#endif
#endif
}
static __device__ inline void wait_async_le8() {
#if HAVE_ASYNC_LDS
#if __has_builtin(__builtin_amdgcn_s_wait_asynccnt)
    __builtin_amdgcn_s_wait_asynccnt(8);
#else
    asm volatile("s_wait_asynccnt 8" ::: "memory");
#endif
#endif
}

static __device__ inline v8f wmma_f16(v16h a, v16h b, v8f c) {
    return __builtin_amdgcn_wmma_f32_16x16x32_f16(
        /*neg_a=*/false, a, /*neg_b=*/false, b,
        /*c_mod=*/(short)0, c, /*reuse_a=*/false, /*reuse_b=*/false);
}

// A-fragment (16x32, M x K) from row-major LDS tile. base -> (row0, k0).
// Lane group g = lane>>4: lanes hold k = [8g..8g+7] then [16+8g..23+8g],
// row m = lane&15 (ISA 16-bit A layout). ld must be a multiple of 8.
static __device__ inline v16h load_frag_a(const _Float16* base, int ld) {
    const int lane = threadIdx.x & 31;
    const int g = lane >> 4, m = lane & 15;
    const _Float16* p = base + m * ld + 8 * g;
    v8h lo = *(const v8h*)(p);
    v8h hi = *(const v8h*)(p + 16);
    v16h r;
#pragma unroll
    for (int t = 0; t < 8; ++t) { r[t] = lo[t]; r[t + 8] = hi[t]; }
    return r;
}

// B-fragment (32x16, K x N) from an [N][K]-major LDS tile. base -> (n0 row, k0).
// Lane group g holds k = [8g..8g+7] then [16+8g..23+8g], column n = lane&15.
static __device__ inline v16h load_frag_b_nk(const _Float16* base, int ld) {
    const int lane = threadIdx.x & 31;
    const int g = lane >> 4, n = lane & 15;
    const _Float16* p = base + n * ld + 8 * g;
    v8h lo = *(const v8h*)(p);
    v8h hi = *(const v8h*)(p + 16);
    v16h r;
#pragma unroll
    for (int t = 0; t < 8; ++t) { r[t] = lo[t]; r[t + 8] = hi[t]; }
    return r;
}

// ---------------------------------------------------------------------------
// Kernel 1: projection GEMM.  f16 dst = src[M,1024] @ W[1024,128].
// which 0/1 (Q/K): dst layout [B*T][128].  which 2 (V): TRANSPOSED [B][128][T]
// so the attention kernel can stage V tiles with contiguous 16B async copies.
// grid = (M/128, 3), block = 256 (8 waves). f32->f16 via v_cvt_pk_rz_f16_f32.
// ---------------------------------------------------------------------------
__global__ __launch_bounds__(256) void proj_kernel(
    const float* __restrict__ q, const float* __restrict__ k,
    const float* __restrict__ v,
    const float* __restrict__ Wq, const float* __restrict__ Wk,
    const float* __restrict__ Wv,
    _Float16* __restrict__ Qf, _Float16* __restrict__ Kf,
    _Float16* __restrict__ Vt)
{
    const int which = blockIdx.y;
    const float* __restrict__ src = (which == 0) ? q : (which == 1) ? k : v;
    const float* __restrict__ W   = (which == 0) ? Wq : (which == 1) ? Wk : Wv;

    const int row0 = blockIdx.x * 128;

    // pool reused: [a_tile 128x40 | w_tile 128x40] during GEMM, tbuf 128x136 after
    __shared__ __align__(16) _Float16 pool[20480];
    _Float16* a_tile = pool;              // [row][k], ld 40
    _Float16* w_tile = pool + 128 * 40;   // [n][k],  ld 40
    _Float16* tbuf   = pool;              // epilogue restage, ld 136

    const int tid  = threadIdx.x;
    const int lane = tid & 31;
    const int wave = tid >> 5;
    const int g = lane >> 4, ln = lane & 15;

    v8f acc[8] = {};

    for (int kc = 0; kc < CDIM; kc += 32) {
        // Stage A: 128 rows x 32 k (float4 loads, packed-convert, b32 stores)
#pragma unroll
        for (int it = 0; it < 4; ++it) {
            int idx = tid + it * 256;          // 0..1023
            int r   = idx >> 3;                // row 0..127
            int c4  = (idx & 7) * 4;           // k   0..28
            float4 f = *(const float4*)(src + (size_t)(row0 + r) * CDIM + kc + c4);
            _Float16* s = a_tile + r * 40 + c4;
            *(v2hp*)(s + 0) = __builtin_amdgcn_cvt_pkrtz(f.x, f.y);
            *(v2hp*)(s + 2) = __builtin_amdgcn_cvt_pkrtz(f.z, f.w);
        }
        // Stage W transposed [n][k]: each thread covers 2 adjacent k rows x 4 n,
        // so packed (k,k+1) pairs are contiguous in the [n][k] tile -> b32 stores.
#pragma unroll
        for (int it = 0; it < 2; ++it) {
            int idx = tid + it * 256;          // 0..511
            int kk2 = idx >> 5;                // k pair 0..15
            int n4  = (idx & 31) * 4;          // n 0..124
            float4 f0 = *(const float4*)(W + (size_t)(kc + 2 * kk2 + 0) * HSD + n4);
            float4 f1 = *(const float4*)(W + (size_t)(kc + 2 * kk2 + 1) * HSD + n4);
            *(v2hp*)(w_tile + (n4 + 0) * 40 + 2 * kk2) = __builtin_amdgcn_cvt_pkrtz(f0.x, f1.x);
            *(v2hp*)(w_tile + (n4 + 1) * 40 + 2 * kk2) = __builtin_amdgcn_cvt_pkrtz(f0.y, f1.y);
            *(v2hp*)(w_tile + (n4 + 2) * 40 + 2 * kk2) = __builtin_amdgcn_cvt_pkrtz(f0.z, f1.z);
            *(v2hp*)(w_tile + (n4 + 3) * 40 + 2 * kk2) = __builtin_amdgcn_cvt_pkrtz(f0.w, f1.w);
        }
        __syncthreads();

        v16h a = load_frag_a(a_tile + (16 * wave) * 40, 40);
#pragma unroll
        for (int nt = 0; nt < 8; ++nt) {
            v16h b = load_frag_b_nk(w_tile + (16 * nt) * 40, 40);
            acc[nt] = wmma_f16(a, b, acc[nt]);
        }
        __syncthreads();
    }

    // Epilogue: restage through LDS so global stores are 16B vectors.
    // C/D layout: acc[nt][r] = (M = 16*wave + r + 8g, N = nt*16 + ln)
    if (which == 2) {
#pragma unroll
        for (int nt = 0; nt < 8; ++nt)
#pragma unroll
            for (int r = 0; r < 8; ++r)
                tbuf[(nt * 16 + ln) * 136 + (16 * wave + r + 8 * g)] =
                    (_Float16)acc[nt][r];                 // [o][t_local]
    } else {
#pragma unroll
        for (int nt = 0; nt < 8; ++nt)
#pragma unroll
            for (int r = 0; r < 8; ++r)
                tbuf[(16 * wave + r + 8 * g) * 136 + nt * 16 + ln] =
                    (_Float16)acc[nt][r];                 // [row][col]
    }
    __syncthreads();

    const int bb = row0 >> 11;            // batch (T = 2048)
    const int t0 = row0 & (TLEN - 1);
    _Float16* dst = (which == 0) ? Qf : (which == 1) ? Kf : Vt;
#pragma unroll
    for (int it = 0; it < 8; ++it) {
        int idx = tid + it * 256;         // 0..2047
        int rr  = idx >> 4;               // 0..127
        int c8  = (idx & 15) * 8;
        v8h d = *(const v8h*)(tbuf + rr * 136 + c8);
        if (which == 2) {
            *(v8h*)(dst + (size_t)bb * HSD * TLEN + (size_t)rr * TLEN + t0 + c8) = d;
        } else {
            *(v8h*)(dst + (size_t)(row0 + rr) * HSD + c8) = d;
        }
    }
}

// ---------------------------------------------------------------------------
// Attention staging helpers (8 async ops per wave per K/V stage).
// ---------------------------------------------------------------------------
static __device__ inline void stage_kv(const _Float16* __restrict__ Kf,
                                       const _Float16* __restrict__ Vt,
                                       size_t base, size_t vbase, int kb,
                                       _Float16* kt, _Float16* vt, int tid) {
    // K tile [key][h]: 64 rows x 16 chunks = 1024 -> 4 per thread
#pragma unroll
    for (int it = 0; it < 4; ++it) {
        int idx = tid + it * 256;
        int r   = idx >> 4;                   // key 0..63
        int c8  = (idx & 15) * 8;             // h 0..120
        const _Float16* gp = Kf + base + (size_t)(kb * KBLK + r) * HSD + c8;
        _Float16*       lp = kt + r * 136 + c8;
#if HAVE_ASYNC_LDS
        async_copy16(gp, lp);
#else
        *(v8h*)lp = *(const v8h*)gp;
#endif
    }
    // Vt tile [o][key]: 128 rows x 8 chunks = 1024 -> 4 per thread
#pragma unroll
    for (int it = 0; it < 4; ++it) {
        int idx = tid + it * 256;
        int o   = idx >> 3;                   // 0..127
        int c8  = (idx & 7) * 8;              // key 0..56
        const _Float16* gp = Vt + vbase + (size_t)o * TLEN + kb * KBLK + c8;
        _Float16*       lp = vt + o * 72 + c8;
#if HAVE_ASYNC_LDS
        async_copy16(gp, lp);
#else
        *(v8h*)lp = *(const v8h*)gp;
#endif
    }
}

// ---------------------------------------------------------------------------
// Kernel 2: causal flash attention (mask j <= i+1, faithful to reference).
// grid = (T/QBLK, B), block = 256 (8 waves). Wave w owns query rows 16w..16w+15.
// Double-buffered async-to-LDS staging; online softmax; P restaged per-wave.
// LDS: Q 34816 + 2xK 34816 + 2xVt 36864 + P 18432 = 124928 B (WGP has 320 KB).
// ---------------------------------------------------------------------------
__global__ __launch_bounds__(256) void attn_kernel(
    const _Float16* __restrict__ Qf, const _Float16* __restrict__ Kf,
    const _Float16* __restrict__ Vt, float* __restrict__ out)
{
    const int b     = blockIdx.y;
    const int qb    = blockIdx.x;
    const int qrow0 = qb * QBLK;
    const size_t base  = (size_t)b * TLEN * HSD;   // Qf/Kf/out base
    const size_t vbase = (size_t)b * HSD * TLEN;   // Vt base

    __shared__ __align__(16) _Float16 q_tile[QBLK * 136];      // [row][h]
    __shared__ __align__(16) _Float16 k_tile[2][KBLK * 136];   // [key][h]
    __shared__ __align__(16) _Float16 vt_tile[2][HSD * 72];    // [o][key]
    __shared__ __align__(16) _Float16 p_tile[8 * 16 * 72];     // per-wave P stage

    const int tid  = threadIdx.x;
    const int lane = tid & 31;
    const int wave = tid >> 5;
    const int g    = lane >> 4;
    const int ln   = lane & 15;

    // Stage Q tile (128 rows x 128 h): 8 x 16B per lane
#pragma unroll
    for (int it = 0; it < 8; ++it) {
        int idx = tid + it * 256;             // 0..2047
        int r   = idx >> 4;                   // row 0..127
        int c8  = (idx & 15) * 8;             // h 0..120
        const _Float16* gp = Qf + base + (size_t)(qrow0 + r) * HSD + c8;
        _Float16*       lp = q_tile + r * 136 + c8;
#if HAVE_ASYNC_LDS
        async_copy16(gp, lp);
#else
        *(v8h*)lp = *(const v8h*)gp;
#endif
    }

    v8f acc_o[8] = {};
    float mrow[8], lrow[8];
#pragma unroll
    for (int r = 0; r < 8; ++r) { mrow[r] = -__builtin_inff(); lrow[r] = 0.f; }

    const float scale = 0.08838834764831845f;  // 1/sqrt(128)
    const int nkb = TLEN / KBLK;
    int kb_end = (qrow0 + QBLK) / KBLK;        // j <= i+1 spills one elem right
    if (kb_end > nkb - 1) kb_end = nkb - 1;

    // prologue: stage kb=0 into buffer 0
    stage_kv(Kf, Vt, base, vbase, 0, k_tile[0], vt_tile[0], tid);

    for (int kb = 0; kb <= kb_end; ++kb) {
        const int cur = kb & 1;
        __syncthreads();   // all waves done reading buffer (1-cur) from kb-1
        if (kb < kb_end) {
            stage_kv(Kf, Vt, base, vbase, kb + 1, k_tile[1 - cur], vt_tile[1 - cur], tid);
            wait_async_le8();   // 8 next-stage ops in flight => stage kb complete
        } else {
            wait_async_le0();
        }
        __syncthreads();

        const _Float16* kt = k_tile[cur];
        const _Float16* vt = vt_tile[cur];

        // S = Q @ K^T for this wave's 16 rows x 64 keys
        v8f s[4] = {};
#pragma unroll
        for (int kc = 0; kc < HSD; kc += 32) {
            v16h a = load_frag_a(q_tile + (16 * wave) * 136 + kc, 136);
#pragma unroll
            for (int nt = 0; nt < 4; ++nt) {
                v16h bf = load_frag_b_nk(kt + (16 * nt) * 136 + kc, 136);
                s[nt] = wmma_f16(a, bf, s[nt]);
            }
        }

        // scale + causal mask (j <= i+1) + row max
        float rmax[8];
#pragma unroll
        for (int r = 0; r < 8; ++r) rmax[r] = -__builtin_inff();
#pragma unroll
        for (int nt = 0; nt < 4; ++nt)
#pragma unroll
            for (int r = 0; r < 8; ++r) {
                int i_g = qrow0 + 16 * wave + r + 8 * g;
                int j_g = kb * KBLK + nt * 16 + ln;
                float val = s[nt][r] * scale;
                val = (j_g <= i_g + 1) ? val : -__builtin_inff();
                s[nt][r] = val;
                rmax[r]  = fmaxf(rmax[r], val);
            }
#pragma unroll
        for (int r = 0; r < 8; ++r) {
            float v0 = rmax[r];
            v0 = fmaxf(v0, __shfl_xor(v0, 1, 32));
            v0 = fmaxf(v0, __shfl_xor(v0, 2, 32));
            v0 = fmaxf(v0, __shfl_xor(v0, 4, 32));
            v0 = fmaxf(v0, __shfl_xor(v0, 8, 32));
            rmax[r] = v0;
        }

        // online softmax update
        float corr[8];
#pragma unroll
        for (int r = 0; r < 8; ++r) {
            float mnew = fmaxf(mrow[r], rmax[r]);
            corr[r] = (mrow[r] == -__builtin_inff()) ? 0.f : __expf(mrow[r] - mnew);
            mrow[r] = mnew;
        }
        float rsum[8];
#pragma unroll
        for (int r = 0; r < 8; ++r) rsum[r] = 0.f;
#pragma unroll
        for (int nt = 0; nt < 4; ++nt)
#pragma unroll
            for (int r = 0; r < 8; ++r) {
                float p = __expf(s[nt][r] - mrow[r]);   // -inf -> 0
                s[nt][r] = p;
                rsum[r] += p;
            }
#pragma unroll
        for (int r = 0; r < 8; ++r) {
            float v0 = rsum[r];
            v0 += __shfl_xor(v0, 1, 32);
            v0 += __shfl_xor(v0, 2, 32);
            v0 += __shfl_xor(v0, 4, 32);
            v0 += __shfl_xor(v0, 8, 32);
            lrow[r] = lrow[r] * corr[r] + v0;
        }
#pragma unroll
        for (int ot = 0; ot < 8; ++ot)
#pragma unroll
            for (int r = 0; r < 8; ++r) acc_o[ot][r] *= corr[r];

        // restage P (C/D layout -> row-major LDS) in this wave's private region
        _Float16* pw = p_tile + wave * 16 * 72;
#pragma unroll
        for (int nt = 0; nt < 4; ++nt)
#pragma unroll
            for (int r = 0; r < 8; ++r)
                pw[(r + 8 * g) * 72 + nt * 16 + ln] = (_Float16)s[nt][r];
        asm volatile("s_wait_dscnt 0" ::: "memory");   // same-wave LDS RAW

        // acc_o += P @ V
#pragma unroll
        for (int kc = 0; kc < KBLK; kc += 32) {
            v16h a = load_frag_a(pw + kc, 72);
#pragma unroll
            for (int ot = 0; ot < 8; ++ot) {
                v16h bf = load_frag_b_nk(vt + (16 * ot) * 72 + kc, 72);
                acc_o[ot] = wmma_f16(a, bf, acc_o[ot]);
            }
        }
    }

    // epilogue: divide by l, store f32
#pragma unroll
    for (int ot = 0; ot < 8; ++ot)
#pragma unroll
        for (int r = 0; r < 8; ++r) {
            int m = qrow0 + 16 * wave + r + 8 * g;
            out[base + (size_t)m * HSD + ot * 16 + ln] = acc_o[ot][r] / lrow[r];
        }
}

// ---------------------------------------------------------------------------
extern "C" void kernel_launch(void* const* d_in, const int* in_sizes, int n_in,
                              void* d_out, int out_size, void* d_ws, size_t ws_size,
                              hipStream_t stream) {
    const float* q  = (const float*)d_in[0];
    const float* k  = (const float*)d_in[1];
    const float* v  = (const float*)d_in[2];
    const float* Wq = (const float*)d_in[3];
    const float* Wk = (const float*)d_in[4];
    const float* Wv = (const float*)d_in[5];
    float* out = (float*)d_out;

    const size_t M = (size_t)BATCH * TLEN;   // 16384
    _Float16* Qf = (_Float16*)d_ws;          // needs 3 * M * 128 * 2 = 12 MB
    _Float16* Kf = Qf + M * HSD;
    _Float16* Vt = Kf + M * HSD;             // [B][128][T] (transposed V)

    dim3 pg((unsigned)(M / 128), 3);
    proj_kernel<<<pg, 256, 0, stream>>>(q, k, v, Wq, Wk, Wv, Qf, Kf, Vt);

    dim3 ag(TLEN / QBLK, BATCH);
    attn_kernel<<<ag, 256, 0, stream>>>(Qf, Kf, Vt, out);
}
